// LSTM_Block_21930103013492
// MI455X (gfx1250) — compile-verified, matrix operand order checked by reading, not run on hardware
//
#include <hip/hip_runtime.h>
#include <stdint.h>
#include <stddef.h>

// ---------------------------------------------------------------------------
// Types for WMMA (CDNA5 / gfx1250, wave32)
// ---------------------------------------------------------------------------
typedef __attribute__((ext_vector_type(16))) __bf16    v16bf;
typedef __attribute__((ext_vector_type(8)))  float     v8f;
typedef __attribute__((ext_vector_type(4)))  uint32_t  u32x4;

#define LSTM_B  512
#define LSTM_T  168
#define LSTM_I  64
#define LSTM_H  256
#define LSTM_4H 1024

union FragU {
    u32x4 q[2];
    v16bf v;
};

__device__ __forceinline__ uint16_t f2bf(float f) {
    uint32_t u = __float_as_uint(f);
    uint32_t r = u + 0x7fffu + ((u >> 16) & 1u);   // round-to-nearest-even
    return (uint16_t)(r >> 16);
}

__device__ __forceinline__ float sigmoid_f(float x) {
    return 1.0f / (1.0f + __expf(-x));
}
__device__ __forceinline__ float tanh_f(float x) {
    float e = __expf(2.0f * x);
    return (e - 1.0f) / (e + 1.0f);
}

__device__ __forceinline__ v8f wmma_bf16(v16bf a, v16bf b, v8f c) {
    // (neg_a, A, neg_b, B, c_mod, C, reuse_a, reuse_b)
    return __builtin_amdgcn_wmma_f32_16x16x32_bf16(
        false, a, false, b, (short)0, c, false, false);
}

__device__ __forceinline__ v16bf gload_frag(const uint16_t* row, int k0) {
    FragU u;
    u.q[0] = *(const u32x4*)(row + k0);
    u.q[1] = *(const u32x4*)(row + k0 + 16);
    return u.v;
}

// ---------------------------------------------------------------------------
// Prep kernels: f32 -> bf16 conversion, bias folding (b_ih + b_hh)
// ---------------------------------------------------------------------------
__global__ void f32_to_bf16_kernel(const float* __restrict__ src,
                                   uint16_t* __restrict__ dst, int n) {
    int i = blockIdx.x * blockDim.x + threadIdx.x;
    if (i < n) dst[i] = f2bf(src[i]);
}

__global__ void bias_combine_kernel(const float* __restrict__ a,
                                    const float* __restrict__ b,
                                    float* __restrict__ o, int n) {
    int i = blockIdx.x * blockDim.x + threadIdx.x;
    if (i < n) o[i] = a[i] + b[i];
}

// ---------------------------------------------------------------------------
// Fused 2-layer LSTM. Grid = B/16 = 32 blocks, 512 threads (16 waves).
// Each block owns 16 batch rows for the whole sequence (both layers).
// Wave w owns output tiles n = 256*g + 16*w for g=0..3 (gates i,f,g,o):
// acc[g] holds gate g for hidden column j = 16w + (lane&15), so the LSTM
// pointwise update happens entirely in registers; cell state is resident.
// Weight fragments stream from L2 (1.7 MB bf16, L2-resident) with a
// sched_barrier-enforced 2-stage pipeline + cross-stage chunk-0 prefetch.
// ---------------------------------------------------------------------------
__global__ __launch_bounds__(512) void lstm2_fused_kernel(
    const uint16_t* __restrict__ xbf,     // [B,T,I] bf16
    const uint16_t* __restrict__ Wih1,    // [4H, I] bf16
    const uint16_t* __restrict__ Whh1,    // [4H, H] bf16
    const uint16_t* __restrict__ Wih2,    // [4H, H] bf16
    const uint16_t* __restrict__ Whh2,    // [4H, H] bf16
    const float*    __restrict__ bias1,   // [4H] = b_ih1 + b_hh1
    const float*    __restrict__ bias2,   // [4H]
    float*          __restrict__ out)     // [B,T,H] f32
{
    __shared__ __align__(16) uint16_t h1s[16 * LSTM_H];   // bf16 h of layer 1
    __shared__ __align__(16) uint16_t h2s[16 * LSTM_H];   // bf16 h of layer 2

    const int tid    = threadIdx.x;
    const int wave   = tid >> 5;        // 0..15
    const int lane   = tid & 31;
    const int lhalf  = lane >> 4;       // 0: lanes 0-15, 1: lanes 16-31
    const int l15    = lane & 15;
    const int batch0 = blockIdx.x * 16;
    const int kloff  = lhalf * 8;       // A/B fragment K phase per half-wave

    for (int i = tid; i < 16 * LSTM_H; i += 512) { h1s[i] = 0; h2s[i] = 0; }
    __syncthreads();

    const int jcol = wave * 16 + l15;   // this lane's hidden column (0..255)

    float b1g[4], b2g[4];
#pragma unroll
    for (int g = 0; g < 4; ++g) {
        b1g[g] = bias1[g * 256 + jcol];
        b2g[g] = bias2[g * 256 + jcol];
    }

    float c1[8], c2[8];
#pragma unroll
    for (int r = 0; r < 8; ++r) { c1[r] = 0.0f; c2[r] = 0.0f; }

    // Per-lane weight-row pointers (B fragments), n = 256*g + jcol.
    const uint16_t* whh1r[4]; const uint16_t* wih1r[4];
    const uint16_t* whh2r[4]; const uint16_t* wih2r[4];
#pragma unroll
    for (int g = 0; g < 4; ++g) {
        const size_t n = (size_t)(g * 256 + jcol);
        whh1r[g] = Whh1 + n * LSTM_H;
        wih1r[g] = Wih1 + n * LSTM_I;
        whh2r[g] = Whh2 + n * LSTM_H;
        wih2r[g] = Wih2 + n * LSTM_H;
    }

    const uint16_t* xrow = xbf + (size_t)(batch0 + l15) * LSTM_T * LSTM_I;
    float* orow = out + (size_t)batch0 * LSTM_T * LSTM_H;

    const v8f vzero = {0.f, 0.f, 0.f, 0.f, 0.f, 0.f, 0.f, 0.f};

    // Cross-stage preloaded chunk-0 fragments of the next GEMM.
    FragU aP; v16bf bP[4];
    {   // initial preload: layer-1 chunk 0 of t=0 (h1s is zeroed + synced)
        const int k0 = kloff;
        aP.q[0] = *(const u32x4*)(&h1s[l15 * LSTM_H + k0]);
        aP.q[1] = *(const u32x4*)(&h1s[l15 * LSTM_H + k0 + 16]);
#pragma unroll
        for (int g = 0; g < 4; ++g) bP[g] = gload_frag(whh1r[g], k0);
    }

#pragma unroll 1
    for (int t = 0; t < LSTM_T; ++t) {
        // Opaque zero offset, refreshed each iteration: keeps weight loads
        // inside the loop (global AS preserved, no hoisting/spilling).
        int z = 0;
        asm volatile("" : "+v"(z));

        const uint16_t* xt = xrow + t * LSTM_I;

        auto load_l1 = [&](int kc, FragU& a, v16bf (&b)[4]) {
            if (kc < 8) {                       // recurrent half: h1 @ Whh1^T
                const int k0 = kc * 32 + kloff;
                a.q[0] = *(const u32x4*)(&h1s[l15 * LSTM_H + k0]);
                a.q[1] = *(const u32x4*)(&h1s[l15 * LSTM_H + k0 + 16]);
#pragma unroll
                for (int g = 0; g < 4; ++g) b[g] = gload_frag(whh1r[g] + z, k0);
            } else {                            // input half: x_t @ Wih1^T
                const int k0 = (kc - 8) * 32 + kloff;
                a.v = gload_frag(xt + z, k0);
#pragma unroll
                for (int g = 0; g < 4; ++g) b[g] = gload_frag(wih1r[g] + z, k0);
            }
        };
        auto load_l2 = [&](int kc, FragU& a, v16bf (&b)[4]) {
            if (kc < 8) {                       // recurrent half: h2 @ Whh2^T
                const int k0 = kc * 32 + kloff;
                a.q[0] = *(const u32x4*)(&h2s[l15 * LSTM_H + k0]);
                a.q[1] = *(const u32x4*)(&h2s[l15 * LSTM_H + k0 + 16]);
#pragma unroll
                for (int g = 0; g < 4; ++g) b[g] = gload_frag(whh2r[g] + z, k0);
            } else {                            // input half: h1(t) @ Wih2^T
                const int k0 = (kc - 8) * 32 + kloff;
                a.q[0] = *(const u32x4*)(&h1s[l15 * LSTM_H + k0]);
                a.q[1] = *(const u32x4*)(&h1s[l15 * LSTM_H + k0 + 16]);
#pragma unroll
                for (int g = 0; g < 4; ++g) b[g] = gload_frag(wih2r[g] + z, k0);
            }
        };

        // ================= layer 1: g = h1 @ Whh1^T + x_t @ Wih1^T ==========
        v8f acc[4];
#pragma unroll
        for (int g = 0; g < 4; ++g) acc[g] = vzero;

        {
            FragU a[2]; v16bf b[2][4];
            a[0] = aP;
#pragma unroll
            for (int g = 0; g < 4; ++g) b[0][g] = bP[g];
#pragma unroll
            for (int kc = 0; kc < 10; ++kc) {   // 8 recurrent + 2 input chunks
                const int cur = kc & 1, nxt = cur ^ 1;
                if (kc + 1 < 10) load_l1(kc + 1, a[nxt], b[nxt]);
                __builtin_amdgcn_sched_barrier(0);   // keep prefetch above MMAs
                acc[0] = wmma_bf16(a[cur].v, b[cur][0], acc[0]);
                acc[1] = wmma_bf16(a[cur].v, b[cur][1], acc[1]);
                acc[2] = wmma_bf16(a[cur].v, b[cur][2], acc[2]);
                acc[3] = wmma_bf16(a[cur].v, b[cur][3], acc[3]);
                __builtin_amdgcn_sched_barrier(0);
            }
        }
        __syncthreads();                 // all waves done reading h1s

        // Cross-stage prefetch: layer-2 chunk 0 (h2s & Whh2 are stable here).
        load_l2(0, aP, bP);
        __builtin_amdgcn_sched_barrier(0);

        // ---- pointwise LSTM update (layer 1), all in registers -------------
#pragma unroll
        for (int r = 0; r < 8; ++r) {
            float ig = sigmoid_f(acc[0][r] + b1g[0]);
            float fg = sigmoid_f(acc[1][r] + b1g[1]);
            float gg = tanh_f   (acc[2][r] + b1g[2]);
            float og = sigmoid_f(acc[3][r] + b1g[3]);
            float cn = fg * c1[r] + ig * gg;
            c1[r] = cn;
            float h  = og * tanh_f(cn);
            h1s[(r + 8 * lhalf) * LSTM_H + jcol] = f2bf(h);
        }
        __syncthreads();                 // h1(t) visible to all waves

        // ================= layer 2: g = h2 @ Whh2^T + h1 @ Wih2^T ==========
#pragma unroll
        for (int g = 0; g < 4; ++g) acc[g] = vzero;

        {
            FragU a[2]; v16bf b[2][4];
            a[0] = aP;
#pragma unroll
            for (int g = 0; g < 4; ++g) b[0][g] = bP[g];
#pragma unroll
            for (int kc = 0; kc < 16; ++kc) {   // 8 recurrent + 8 input chunks
                const int cur = kc & 1, nxt = cur ^ 1;
                if (kc + 1 < 16) load_l2(kc + 1, a[nxt], b[nxt]);
                __builtin_amdgcn_sched_barrier(0);   // keep prefetch above MMAs
                acc[0] = wmma_bf16(a[cur].v, b[cur][0], acc[0]);
                acc[1] = wmma_bf16(a[cur].v, b[cur][1], acc[1]);
                acc[2] = wmma_bf16(a[cur].v, b[cur][2], acc[2]);
                acc[3] = wmma_bf16(a[cur].v, b[cur][3], acc[3]);
                __builtin_amdgcn_sched_barrier(0);
            }
        }
        __syncthreads();                 // all waves done reading h2s

        // Cross-stage prefetch: layer-1 chunk 0 for t+1 (h1s(t) & Whh1 stable;
        // act2 below only writes h2s/out).
        load_l1(0, aP, bP);
        __builtin_amdgcn_sched_barrier(0);

        // ---- pointwise LSTM update (layer 2) + output store ----------------
#pragma unroll
        for (int r = 0; r < 8; ++r) {
            float ig = sigmoid_f(acc[0][r] + b2g[0]);
            float fg = sigmoid_f(acc[1][r] + b2g[1]);
            float gg = tanh_f   (acc[2][r] + b2g[2]);
            float og = sigmoid_f(acc[3][r] + b2g[3]);
            float cn = fg * c2[r] + ig * gg;
            c2[r] = cn;
            float h  = og * tanh_f(cn);
            const int m = r + 8 * lhalf;
            h2s[m * LSTM_H + jcol] = f2bf(h);
            orow[(size_t)m * LSTM_T * LSTM_H + (size_t)t * LSTM_H + jcol] = h;
        }
        __syncthreads();                 // h2(t) writes ordered before next read
    }
}

// ---------------------------------------------------------------------------
// Host-side launch
// ---------------------------------------------------------------------------
extern "C" void kernel_launch(void* const* d_in, const int* in_sizes, int n_in,
                              void* d_out, int out_size, void* d_ws, size_t ws_size,
                              hipStream_t stream) {
    (void)in_sizes; (void)n_in; (void)out_size; (void)ws_size;

    const float* x     = (const float*)d_in[0];
    const float* Wih1f = (const float*)d_in[1];
    const float* Whh1f = (const float*)d_in[2];
    const float* bih1  = (const float*)d_in[3];
    const float* bhh1  = (const float*)d_in[4];
    const float* Wih2f = (const float*)d_in[5];
    const float* Whh2f = (const float*)d_in[6];
    const float* bih2  = (const float*)d_in[7];
    const float* bhh2  = (const float*)d_in[8];

    const int nWih1 = LSTM_4H * LSTM_I;          // 65536
    const int nWhh  = LSTM_4H * LSTM_H;          // 262144
    const int nX    = LSTM_B * LSTM_T * LSTM_I;  // 5505024

    uint8_t* ws = (uint8_t*)d_ws;
    size_t off = 0;
    uint16_t* wih1 = (uint16_t*)(ws + off); off += (size_t)nWih1 * 2;
    uint16_t* whh1 = (uint16_t*)(ws + off); off += (size_t)nWhh  * 2;
    uint16_t* wih2 = (uint16_t*)(ws + off); off += (size_t)nWhh  * 2;
    uint16_t* whh2 = (uint16_t*)(ws + off); off += (size_t)nWhh  * 2;
    float*    b1   = (float*)   (ws + off); off += LSTM_4H * sizeof(float);
    float*    b2   = (float*)   (ws + off); off += LSTM_4H * sizeof(float);
    uint16_t* xbf  = (uint16_t*)(ws + off); off += (size_t)nX * 2;

    const int TB = 256;
    f32_to_bf16_kernel<<<(nWih1 + TB - 1) / TB, TB, 0, stream>>>(Wih1f, wih1, nWih1);
    f32_to_bf16_kernel<<<(nWhh  + TB - 1) / TB, TB, 0, stream>>>(Whh1f, whh1, nWhh);
    f32_to_bf16_kernel<<<(nWhh  + TB - 1) / TB, TB, 0, stream>>>(Wih2f, wih2, nWhh);
    f32_to_bf16_kernel<<<(nWhh  + TB - 1) / TB, TB, 0, stream>>>(Whh2f, whh2, nWhh);
    f32_to_bf16_kernel<<<(nX    + TB - 1) / TB, TB, 0, stream>>>(x,     xbf,  nX);
    bias_combine_kernel<<<(LSTM_4H + TB - 1) / TB, TB, 0, stream>>>(bih1, bhh1, b1, LSTM_4H);
    bias_combine_kernel<<<(LSTM_4H + TB - 1) / TB, TB, 0, stream>>>(bih2, bhh2, b2, LSTM_4H);

    lstm2_fused_kernel<<<LSTM_B / 16, 512, 0, stream>>>(
        xbf, wih1, whh1, wih2, whh2, b1, b2, (float*)d_out);
}
